// TransformerEncoderLayerWithMoE_14972255994100
// MI455X (gfx1250) — matrix operand order repaired and checked by reading, not compiled, not run on hardware
//
#include <hip/hip_runtime.h>

#define T_TOK   4096
#define DMODEL  1024
#define NHEAD   16
#define DH      64
#define DFF     4096
#define N_EXP   8
#define TOPK    2
#define SEQ     1024
#define NBATCH  4

typedef __attribute__((ext_vector_type(16))) __bf16 v16bf;
typedef __attribute__((ext_vector_type(8)))  float  v8f;
typedef unsigned short ushort_t;

static __device__ inline ushort_t f2b(float f) {
  union { __bf16 h; ushort_t u; } c;
  c.h = (__bf16)f;               // native v_cvt (RNE) on gfx1250
  return c.u;
}
static __device__ inline __bf16 b2h(ushort_t s) {
  union { ushort_t u; __bf16 h; } c; c.u = s; return c.h;
}
static __device__ inline v8f v8zero() {
  v8f z = {0.f,0.f,0.f,0.f,0.f,0.f,0.f,0.f};
  return z;
}
static __device__ inline uint4 pack8(float4 a, float4 b) {
  union { ushort_t u[8]; uint4 v; } p;
  p.u[0] = f2b(a.x); p.u[1] = f2b(a.y); p.u[2] = f2b(a.z); p.u[3] = f2b(a.w);
  p.u[4] = f2b(b.x); p.u[5] = f2b(b.y); p.u[6] = f2b(b.z); p.u[7] = f2b(b.w);
  return p.v;
}

// ---------------------------------------------------------------------------
// fp32 -> bf16 (8 elements/thread, vectorized)
// ---------------------------------------------------------------------------
__global__ void cvt_bf16(const float* __restrict__ in, ushort_t* __restrict__ out,
                         long long n8) {
  long long i = (long long)blockIdx.x * blockDim.x + threadIdx.x;
  if (i >= n8) return;
  const float4* ip = (const float4*)(in + i * 8);
  ((uint4*)(out + i * 8))[0] = pack8(ip[0], ip[1]);
}

// fp32 [E][K][N] -> bf16 [E][N][K] (transpose so GEMM B-tiles are K-contiguous)
__global__ void cvt_bf16_t(const float* __restrict__ in, ushort_t* __restrict__ out,
                           int K, int N) {
  long long i = (long long)blockIdx.x * blockDim.x + threadIdx.x;
  long long per = (long long)K * N;
  long long total = (long long)N_EXP * per;
  if (i >= total) return;
  int e = (int)(i / per);
  long long rem = i - (long long)e * per;
  int n = (int)(rem / K);
  int k = (int)(rem - (long long)n * K);
  out[(long long)e * per + (long long)n * K + k] =
      f2b(in[(long long)e * per + (long long)k * N + n]);
}

// ---------------------------------------------------------------------------
// Generic WMMA bf16 GEMM: C[M,N] = act( A[M,K](f32) * Bt[N,K](bf16)^T + bias )
// Block tile 128x128x32, 8 waves (4 along M x 2 along N), each wave 32x64.
// Per-expert base/count (blockIdx.z = expert) and optional row-gather on A.
// Each thread owns one (row, 16-col half) of the tile: b128 loads, b128 LDS
// stores, guard + gather hoisted out of the K loop; next tile prefetched.
// ---------------------------------------------------------------------------
__global__ __launch_bounds__(256) void gemm_bf16_wmma(
    const float* __restrict__ A, int lda,
    const ushort_t* __restrict__ Bt, int K, long long bStride,
    const float* __restrict__ bias, int biasStride,
    float* __restrict__ C, int ldc,
    const int* __restrict__ counts, const int* __restrict__ offs,
    const int* __restrict__ rowmap, int Mfixed, int relu)
{
  const int tid = threadIdx.x, lane = tid & 31, w = tid >> 5;
  const int e = blockIdx.z;
  const int base = offs ? offs[e] : 0;
  const int Mcnt = counts ? counts[e] : Mfixed;
  const int mbase = blockIdx.y * 128;
  if (mbase >= Mcnt) return;
  const int nbase = blockIdx.x * 128;
  const ushort_t* Bte = Bt + (long long)e * bStride;
  const float* biasp = bias + (long long)e * biasStride;

  __shared__ __align__(16) ushort_t As[128][40];
  __shared__ __align__(16) ushort_t Bs[128][40];

  const int wm = w & 3, wn = w >> 2;
  const int hlf = lane >> 4, l16 = lane & 15;

  // loader assignment: thread -> (tile row, col-half)
  const int lr = tid >> 1;
  const int lc = (tid & 1) * 16;
  const float* aptr = nullptr;
  {
    int am = mbase + lr;
    if (am < Mcnt) {
      int arow = rowmap ? rowmap[base + am] : (base + am);
      aptr = A + (long long)arow * lda;
    }
  }
  const ushort_t* bptr = Bte + (long long)(nbase + lr) * K;

  v8f acc[2][4];
  #pragma unroll
  for (int i = 0; i < 2; ++i)
    #pragma unroll
    for (int j = 0; j < 4; ++j)
      acc[i][j] = v8zero();

  for (int kt = 0; kt < K; kt += 32) {
    float4 a0 = make_float4(0.f,0.f,0.f,0.f), a1 = a0, a2 = a0, a3 = a0;
    if (aptr) {
      const float4* ap = (const float4*)(aptr + kt + lc);
      a0 = ap[0]; a1 = ap[1]; a2 = ap[2]; a3 = ap[3];
    }
    const uint4* bp = (const uint4*)(bptr + kt + lc);
    uint4 b0 = bp[0], b1 = bp[1];
    if (kt + 32 < K) {
      if (aptr) __builtin_prefetch(aptr + kt + 32 + lc, 0, 1);
      __builtin_prefetch(bptr + kt + 32 + lc, 0, 1);
    }
    *(uint4*)&As[lr][lc]     = pack8(a0, a1);
    *(uint4*)&As[lr][lc + 8] = pack8(a2, a3);
    *(uint4*)&Bs[lr][lc]     = b0;
    *(uint4*)&Bs[lr][lc + 8] = b1;
    __syncthreads();

    v16bf af[2];
    #pragma unroll
    for (int ms = 0; ms < 2; ++ms) {
      int row = wm * 32 + ms * 16 + l16;
      #pragma unroll
      for (int i = 0; i < 8; ++i) {
        af[ms][i]     = b2h(As[row][hlf * 8 + i]);
        af[ms][8 + i] = b2h(As[row][hlf * 8 + 16 + i]);
      }
    }
    #pragma unroll
    for (int ns = 0; ns < 4; ++ns) {
      v16bf bfr;
      int nrow = wn * 64 + ns * 16 + l16;
      #pragma unroll
      for (int i = 0; i < 8; ++i) {
        bfr[i]     = b2h(Bs[nrow][hlf * 8 + i]);
        bfr[8 + i] = b2h(Bs[nrow][hlf * 8 + 16 + i]);
      }
      #pragma unroll
      for (int ms = 0; ms < 2; ++ms)
        acc[ms][ns] = __builtin_amdgcn_wmma_f32_16x16x32_bf16(
            false, af[ms], false, bfr, (short)0, acc[ms][ns], false, false);
    }
    __syncthreads();
  }

  #pragma unroll
  for (int ms = 0; ms < 2; ++ms) {
    #pragma unroll
    for (int ns = 0; ns < 4; ++ns) {
      int col = nbase + wn * 64 + ns * 16 + l16;
      float bv = bias ? biasp[col] : 0.f;
      #pragma unroll
      for (int r = 0; r < 8; ++r) {
        int m = mbase + wm * 32 + ms * 16 + hlf * 8 + r;
        if (m < Mcnt) {
          float v = acc[ms][ns][r] + bv;
          if (relu) v = fmaxf(v, 0.f);
          C[(long long)(base + m) * ldc + col] = v;
        }
      }
    }
  }
}

// ---------------------------------------------------------------------------
// Flash attention: qkv [T, 3*1024] f32 -> out [T, 1024] f32 (pre out-proj).
// Block = 8 waves handling 128 q-rows of one (b,h); K/V tiles of 64 streamed
// through LDS; online softmax in WMMA C-layout; P restaged via per-wave LDS.
// ---------------------------------------------------------------------------
__global__ __launch_bounds__(256) void flash_attn_kernel(
    const float* __restrict__ qkv, float* __restrict__ out)
{
  const int b = blockIdx.z, h = blockIdx.y, qt = blockIdx.x;
  const int tid = threadIdx.x, lane = tid & 31, w = tid >> 5;
  const int hlf = lane >> 4, l16 = lane & 15;

  __shared__ __align__(16) ushort_t Qs[128][72];
  __shared__ __align__(16) ushort_t Ks[64][72];
  __shared__ __align__(16) ushort_t Vt[64][72];      // transposed: [dh][key]
  __shared__ __align__(16) ushort_t Ps[8][16][72];   // per-wave P staging

  const long long ld = 3LL * DMODEL;
  const long long qbase = ((long long)b * SEQ + (long long)qt * 128) * ld + h * DH;

  // loader assignment: (row, 16-col quarter)
  const int qr = tid >> 2;
  const int qc = (tid & 3) * 16;
  #pragma unroll
  for (int i = 0; i < 2; ++i) {
    int row = qr + i * 64;
    const float4* qp = (const float4*)(qkv + qbase + (long long)row * ld + qc);
    *(uint4*)&Qs[row][qc]     = pack8(qp[0], qp[1]);
    *(uint4*)&Qs[row][qc + 8] = pack8(qp[2], qp[3]);
  }
  __syncthreads();

  const int m0 = w * 16;
  v16bf qa0, qa1;
  {
    int row = m0 + l16;
    #pragma unroll
    for (int i = 0; i < 8; ++i) {
      qa0[i]     = b2h(Qs[row][hlf * 8 + i]);
      qa0[8 + i] = b2h(Qs[row][hlf * 8 + 16 + i]);
      qa1[i]     = b2h(Qs[row][32 + hlf * 8 + i]);
      qa1[8 + i] = b2h(Qs[row][32 + hlf * 8 + 16 + i]);
    }
  }

  v8f o[4];
  #pragma unroll
  for (int i = 0; i < 4; ++i) o[i] = v8zero();
  float mrow[8], lrow[8];
  #pragma unroll
  for (int r = 0; r < 8; ++r) { mrow[r] = -1e30f; lrow[r] = 0.f; }

  const int kr = tid >> 2;         // 0..63
  const int kc = (tid & 3) * 16;   // 0/16/32/48

  for (int j = 0; j < SEQ; j += 64) {
    __syncthreads();
    const long long kb = ((long long)b * SEQ + j) * ld + DMODEL + h * DH;
    const long long vb = ((long long)b * SEQ + j) * ld + 2 * DMODEL + h * DH;
    {
      const float4* kp = (const float4*)(qkv + kb + (long long)kr * ld + kc);
      *(uint4*)&Ks[kr][kc]     = pack8(kp[0], kp[1]);
      *(uint4*)&Ks[kr][kc + 8] = pack8(kp[2], kp[3]);
      const float4* vp = (const float4*)(qkv + vb + (long long)kr * ld + kc);
      union { float4 f4[4]; float s[16]; } vv;
      vv.f4[0] = vp[0]; vv.f4[1] = vp[1]; vv.f4[2] = vp[2]; vv.f4[3] = vp[3];
      #pragma unroll
      for (int jj = 0; jj < 16; ++jj)
        Vt[kc + jj][kr] = f2b(vv.s[jj]);   // transposed scatter
      if (j + 64 < SEQ) {
        __builtin_prefetch(qkv + kb + 64 * ld + (long long)kr * ld + kc, 0, 1);
        __builtin_prefetch(qkv + vb + 64 * ld + (long long)kr * ld + kc, 0, 1);
      }
    }
    __syncthreads();

    v8f sacc[4];
    #pragma unroll
    for (int nt = 0; nt < 4; ++nt) sacc[nt] = v8zero();
    #pragma unroll
    for (int nt = 0; nt < 4; ++nt) {
      v16bf kf0, kf1;
      int nrow = nt * 16 + l16;
      #pragma unroll
      for (int i = 0; i < 8; ++i) {
        kf0[i]     = b2h(Ks[nrow][hlf * 8 + i]);
        kf0[8 + i] = b2h(Ks[nrow][hlf * 8 + 16 + i]);
        kf1[i]     = b2h(Ks[nrow][32 + hlf * 8 + i]);
        kf1[8 + i] = b2h(Ks[nrow][32 + hlf * 8 + 16 + i]);
      }
      sacc[nt] = __builtin_amdgcn_wmma_f32_16x16x32_bf16(
          false, qa0, false, kf0, (short)0, sacc[nt], false, false);
      sacc[nt] = __builtin_amdgcn_wmma_f32_16x16x32_bf16(
          false, qa1, false, kf1, (short)0, sacc[nt], false, false);
    }

    #pragma unroll
    for (int r = 0; r < 8; ++r) {
      float s0 = sacc[0][r] * 0.125f;
      float s1 = sacc[1][r] * 0.125f;
      float s2 = sacc[2][r] * 0.125f;
      float s3 = sacc[3][r] * 0.125f;
      float mx = fmaxf(fmaxf(s0, s1), fmaxf(s2, s3));
      #pragma unroll
      for (int off = 8; off > 0; off >>= 1) mx = fmaxf(mx, __shfl_xor(mx, off));
      float mnew = fmaxf(mrow[r], mx);
      float alpha = __expf(mrow[r] - mnew);
      float p0 = __expf(s0 - mnew), p1 = __expf(s1 - mnew);
      float p2 = __expf(s2 - mnew), p3 = __expf(s3 - mnew);
      float rsum = p0 + p1 + p2 + p3;
      #pragma unroll
      for (int off = 8; off > 0; off >>= 1) rsum += __shfl_xor(rsum, off);
      lrow[r] = lrow[r] * alpha + rsum;
      mrow[r] = mnew;
      #pragma unroll
      for (int nt = 0; nt < 4; ++nt) o[nt][r] *= alpha;
      int prow = hlf * 8 + r;
      Ps[w][prow][0 * 16 + l16] = f2b(p0);
      Ps[w][prow][1 * 16 + l16] = f2b(p1);
      Ps[w][prow][2 * 16 + l16] = f2b(p2);
      Ps[w][prow][3 * 16 + l16] = f2b(p3);
    }

    v16bf pa0, pa1;
    {
      int row = l16;
      #pragma unroll
      for (int i = 0; i < 8; ++i) {
        pa0[i]     = b2h(Ps[w][row][hlf * 8 + i]);
        pa0[8 + i] = b2h(Ps[w][row][hlf * 8 + 16 + i]);
        pa1[i]     = b2h(Ps[w][row][32 + hlf * 8 + i]);
        pa1[8 + i] = b2h(Ps[w][row][32 + hlf * 8 + 16 + i]);
      }
    }
    #pragma unroll
    for (int nt = 0; nt < 4; ++nt) {
      v16bf vf0, vf1;
      int nrow = nt * 16 + l16;
      #pragma unroll
      for (int i = 0; i < 8; ++i) {
        vf0[i]     = b2h(Vt[nrow][hlf * 8 + i]);
        vf0[8 + i] = b2h(Vt[nrow][hlf * 8 + 16 + i]);
        vf1[i]     = b2h(Vt[nrow][32 + hlf * 8 + i]);
        vf1[8 + i] = b2h(Vt[nrow][32 + hlf * 8 + 16 + i]);
      }
      o[nt] = __builtin_amdgcn_wmma_f32_16x16x32_bf16(
          false, pa0, false, vf0, (short)0, o[nt], false, false);
      o[nt] = __builtin_amdgcn_wmma_f32_16x16x32_bf16(
          false, pa1, false, vf1, (short)0, o[nt], false, false);
    }
  }

  #pragma unroll
  for (int r = 0; r < 8; ++r) {
    float inv = 1.f / lrow[r];
    int row = qt * 128 + m0 + hlf * 8 + r;
    long long ob = ((long long)b * SEQ + row) * DMODEL + h * DH;
    #pragma unroll
    for (int nt = 0; nt < 4; ++nt)
      out[ob + nt * 16 + l16] = o[nt][r] * inv;
  }
}

// ---------------------------------------------------------------------------
// y = LayerNorm(a + b) * g + beta  (one block per token row, d=1024)
// ---------------------------------------------------------------------------
__global__ __launch_bounds__(256) void add_ln_kernel(
    const float* __restrict__ a, const float* __restrict__ b,
    const float* __restrict__ g, const float* __restrict__ bt,
    float* __restrict__ y)
{
  int row = blockIdx.x, tid = threadIdx.x, lane = tid & 31, w = tid >> 5;
  __shared__ float rs[8], rs2[8], stats[2];
  const long long rb = (long long)row * DMODEL + tid * 4;
  float4 va = *(const float4*)(a + rb);
  float4 vb = *(const float4*)(b + rb);
  float v[4] = { va.x + vb.x, va.y + vb.y, va.z + vb.z, va.w + vb.w };
  float s = 0.f, s2 = 0.f;
  #pragma unroll
  for (int i = 0; i < 4; ++i) { s += v[i]; s2 += v[i] * v[i]; }
  #pragma unroll
  for (int off = 16; off > 0; off >>= 1) {
    s += __shfl_xor(s, off); s2 += __shfl_xor(s2, off);
  }
  if (lane == 0) { rs[w] = s; rs2[w] = s2; }
  __syncthreads();
  if (tid == 0) {
    float t = 0.f, t2 = 0.f;
    for (int k = 0; k < 8; ++k) { t += rs[k]; t2 += rs2[k]; }
    float mu = t / (float)DMODEL;
    float var = t2 / (float)DMODEL - mu * mu;
    stats[0] = mu; stats[1] = rsqrtf(var + 1e-5f);
  }
  __syncthreads();
  float mu = stats[0], rinv = stats[1];
  const float4 vg = *(const float4*)(g + tid * 4);
  const float4 vt = *(const float4*)(bt + tid * 4);
  float4 r;
  r.x = (v[0] - mu) * rinv * vg.x + vt.x;
  r.y = (v[1] - mu) * rinv * vg.y + vt.y;
  r.z = (v[2] - mu) * rinv * vg.z + vt.z;
  r.w = (v[3] - mu) * rinv * vg.w + vt.w;
  *(float4*)(y + rb) = r;
}

// ---------------------------------------------------------------------------
// Router: one wave per token; softmax over 8 experts, top-2 renormalized.
// ---------------------------------------------------------------------------
__global__ __launch_bounds__(32) void router_kernel(
    const float* __restrict__ x, const float* __restrict__ rw,
    const float* __restrict__ rb,
    int* __restrict__ topi, float* __restrict__ gates,
    int* __restrict__ cnt, float* __restrict__ Psum)
{
  int t = blockIdx.x;
  int lane = threadIdx.x;
  float acc[N_EXP];
  #pragma unroll
  for (int e = 0; e < N_EXP; ++e) acc[e] = 0.f;
  for (int c = lane * 4; c < DMODEL; c += 32 * 4) {
    float4 xv = *(const float4*)(x + (long long)t * DMODEL + c);
    #pragma unroll
    for (int e = 0; e < N_EXP; ++e) {
      float4 wv = *(const float4*)(rw + e * DMODEL + c);
      acc[e] += xv.x * wv.x + xv.y * wv.y + xv.z * wv.z + xv.w * wv.w;
    }
  }
  #pragma unroll
  for (int e = 0; e < N_EXP; ++e)
    #pragma unroll
    for (int off = 16; off > 0; off >>= 1)
      acc[e] += __shfl_xor(acc[e], off);
  if (lane == 0) {
    float p[N_EXP];
    float mx = -1e30f;
    #pragma unroll
    for (int e = 0; e < N_EXP; ++e) { p[e] = acc[e] + rb[e]; mx = fmaxf(mx, p[e]); }
    float s = 0.f;
    #pragma unroll
    for (int e = 0; e < N_EXP; ++e) { p[e] = __expf(p[e] - mx); s += p[e]; }
    float inv = 1.f / s;
    #pragma unroll
    for (int e = 0; e < N_EXP; ++e) p[e] *= inv;
    int i0 = 0;
    #pragma unroll
    for (int e = 1; e < N_EXP; ++e) if (p[e] > p[i0]) i0 = e;
    int i1 = (i0 == 0) ? 1 : 0;
    #pragma unroll
    for (int e = 0; e < N_EXP; ++e) if (e != i0 && p[e] > p[i1]) i1 = e;
    float gs = p[i0] + p[i1];
    topi[t * 2] = i0; topi[t * 2 + 1] = i1;
    gates[t * 2] = p[i0] / gs; gates[t * 2 + 1] = p[i1] / gs;
    atomicAdd(&cnt[i0], 1); atomicAdd(&cnt[i1], 1);
    #pragma unroll
    for (int e = 0; e < N_EXP; ++e) atomicAdd(&Psum[e], p[e]);
  }
}

__global__ void zero_small(int* cnt, float* Psum) {
  int i = threadIdx.x;
  if (i < N_EXP) { cnt[i] = 0; Psum[i] = 0.f; }
}

__global__ void scan_offsets(const int* __restrict__ cnt, int* offs, int* cursor) {
  int o = 0;
  for (int e = 0; e < N_EXP; ++e) { offs[e] = o; cursor[e] = o; o += cnt[e]; }
}

__global__ void scatter_tokens(const int* __restrict__ topi, int* cursor,
                               int* __restrict__ pair_tok, int* __restrict__ pair_pos) {
  int t = blockIdx.x * blockDim.x + threadIdx.x;
  if (t >= T_TOK) return;
  for (int kk = 0; kk < TOPK; ++kk) {
    int e = topi[t * TOPK + kk];
    int pos = atomicAdd(&cursor[e], 1);
    pair_tok[pos] = t;
    pair_pos[t * TOPK + kk] = pos;
  }
}

// Weighted combine of the two expert outputs per token (deterministic).
__global__ __launch_bounds__(256) void combine_kernel(
    const float* __restrict__ ybuf, const float* __restrict__ gates,
    const int* __restrict__ pair_pos, float* __restrict__ moe)
{
  int t = blockIdx.x, tid = threadIdx.x;
  int p0 = pair_pos[t * 2], p1 = pair_pos[t * 2 + 1];
  float g0 = gates[t * 2], g1 = gates[t * 2 + 1];
  float4 y0 = *(const float4*)(ybuf + (long long)p0 * DMODEL + tid * 4);
  float4 y1 = *(const float4*)(ybuf + (long long)p1 * DMODEL + tid * 4);
  float4 r;
  r.x = g0 * y0.x + g1 * y1.x;
  r.y = g0 * y0.y + g1 * y1.y;
  r.z = g0 * y0.z + g1 * y1.z;
  r.w = g0 * y0.w + g1 * y1.w;
  *(float4*)(moe + (long long)t * DMODEL + tid * 4) = r;
}

__global__ void lb_kernel(const int* __restrict__ cnt, const float* __restrict__ Psum,
                          float* __restrict__ out) {
  float lb = 0.f;
  for (int e = 0; e < N_EXP; ++e) {
    float f = (float)cnt[e] / (float)(T_TOK * TOPK);
    float P = Psum[e] / (float)T_TOK;
    lb += f * P;
  }
  *out = (float)N_EXP * lb;
}

// ---------------------------------------------------------------------------
extern "C" void kernel_launch(void* const* d_in, const int* in_sizes, int n_in,
                              void* d_out, int out_size, void* d_ws, size_t ws_size,
                              hipStream_t stream) {
  (void)in_sizes; (void)n_in; (void)out_size; (void)ws_size;
  const float* src        = (const float*)d_in[0];
  const float* in_proj_w  = (const float*)d_in[1];
  const float* in_proj_b  = (const float*)d_in[2];
  const float* out_proj_w = (const float*)d_in[3];
  const float* out_proj_b = (const float*)d_in[4];
  const float* ln1_g = (const float*)d_in[5];
  const float* ln1_b = (const float*)d_in[6];
  const float* router_w = (const float*)d_in[7];
  const float* router_b = (const float*)d_in[8];
  const float* w1 = (const float*)d_in[9];
  const float* b1 = (const float*)d_in[10];
  const float* w2 = (const float*)d_in[11];
  const float* b2 = (const float*)d_in[12];
  const float* ln2_g = (const float*)d_in[13];
  const float* ln2_b = (const float*)d_in[14];

  char* p = (char*)d_ws;
  auto alloc = [&](size_t bytes) -> void* {
    void* r = (void*)p;
    p += (bytes + 255) & ~(size_t)255;
    return r;
  };
  const size_t NPAIR = (size_t)T_TOK * TOPK;  // exactly 8192
  float*    qkv    = (float*)alloc((size_t)T_TOK * 3 * DMODEL * 4);
  float*    attn   = (float*)alloc((size_t)T_TOK * DMODEL * 4);
  float*    attnp  = (float*)alloc((size_t)T_TOK * DMODEL * 4);
  float*    x1     = (float*)alloc((size_t)T_TOK * DMODEL * 4);
  float*    moe    = (float*)alloc((size_t)T_TOK * DMODEL * 4);
  ushort_t* wqkv_b = (ushort_t*)alloc((size_t)3 * DMODEL * DMODEL * 2);
  ushort_t* wo_b   = (ushort_t*)alloc((size_t)DMODEL * DMODEL * 2);
  ushort_t* w1t    = (ushort_t*)alloc((size_t)N_EXP * DMODEL * DFF * 2);
  ushort_t* w2t    = (ushort_t*)alloc((size_t)N_EXP * DFF * DMODEL * 2);
  float*    hbuf   = (float*)alloc(NPAIR * DFF * 4);
  float*    ybuf   = (float*)alloc(NPAIR * DMODEL * 4);
  int*      topi   = (int*)alloc((size_t)T_TOK * TOPK * 4);
  float*    gates  = (float*)alloc((size_t)T_TOK * TOPK * 4);
  int*      cnt    = (int*)alloc(N_EXP * 4);
  int*      offs   = (int*)alloc(N_EXP * 4);
  int*      cursor = (int*)alloc(N_EXP * 4);
  float*    Psum   = (float*)alloc(N_EXP * 4);
  int*      pair_tok = (int*)alloc(NPAIR * 4);
  int*      pair_pos = (int*)alloc((size_t)T_TOK * TOPK * 4);

  // 1) Weight conversions to bf16 ([N,K] layouts)
  cvt_bf16<<<(3 * DMODEL * DMODEL / 8) / 256, 256, 0, stream>>>(
      in_proj_w, wqkv_b, (long long)3 * DMODEL * DMODEL / 8);
  cvt_bf16<<<(DMODEL * DMODEL / 8) / 256, 256, 0, stream>>>(
      out_proj_w, wo_b, (long long)DMODEL * DMODEL / 8);
  cvt_bf16_t<<<(int)(((long long)N_EXP * DMODEL * DFF) / 256), 256, 0, stream>>>(
      w1, w1t, DMODEL, DFF);
  cvt_bf16_t<<<(int)(((long long)N_EXP * DFF * DMODEL) / 256), 256, 0, stream>>>(
      w2, w2t, DFF, DMODEL);

  // 2) QKV projection: [4096,1024] x [1024,3072] + b
  gemm_bf16_wmma<<<dim3(3 * DMODEL / 128, T_TOK / 128, 1), 256, 0, stream>>>(
      src, DMODEL, wqkv_b, DMODEL, 0LL, in_proj_b, 0, qkv, 3 * DMODEL,
      nullptr, nullptr, nullptr, T_TOK, 0);

  // 3) Flash attention
  flash_attn_kernel<<<dim3(SEQ / 128, NHEAD, NBATCH), 256, 0, stream>>>(qkv, attn);

  // 4) Output projection
  gemm_bf16_wmma<<<dim3(DMODEL / 128, T_TOK / 128, 1), 256, 0, stream>>>(
      attn, DMODEL, wo_b, DMODEL, 0LL, out_proj_b, 0, attnp, DMODEL,
      nullptr, nullptr, nullptr, T_TOK, 0);

  // 5) x1 = LN(src + attnp)
  add_ln_kernel<<<T_TOK, 256, 0, stream>>>(src, attnp, ln1_g, ln1_b, x1);

  // 6) Router + top-2 gather lists
  zero_small<<<1, 32, 0, stream>>>(cnt, Psum);
  router_kernel<<<T_TOK, 32, 0, stream>>>(x1, router_w, router_b,
                                          topi, gates, cnt, Psum);
  scan_offsets<<<1, 1, 0, stream>>>(cnt, offs, cursor);
  scatter_tokens<<<T_TOK / 256, 256, 0, stream>>>(topi, cursor, pair_tok, pair_pos);

  // 7) Expert FFN, grouped GEMMs (blocks past per-expert count early-exit)
  gemm_bf16_wmma<<<dim3(DFF / 128, T_TOK / 128, N_EXP), 256, 0, stream>>>(
      x1, DMODEL, w1t, DMODEL, (long long)DFF * DMODEL, b1, DFF, hbuf, DFF,
      cnt, offs, pair_tok, 0, 1 /*relu*/);
  gemm_bf16_wmma<<<dim3(DMODEL / 128, T_TOK / 128, N_EXP), 256, 0, stream>>>(
      hbuf, DFF, w2t, DFF, (long long)DMODEL * DFF, b2, DMODEL, ybuf, DMODEL,
      cnt, offs, nullptr, 0, 0);

  // 8) Combine, final LN, load-balance loss
  combine_kernel<<<T_TOK, 256, 0, stream>>>(ybuf, gates, pair_pos, moe);
  add_ln_kernel<<<T_TOK, 256, 0, stream>>>(x1, moe, ln2_g, ln2_b, (float*)d_out);
  lb_kernel<<<1, 1, 0, stream>>>(cnt, Psum, (float*)d_out + (size_t)T_TOK * DMODEL);
}
